// GraphClassifier_47132971106435
// MI455X (gfx1250) — compile-verified, matrix operand order checked by reading, not run on hardware
//
#include <hip/hip_runtime.h>
#include <hip/hip_bf16.h>
#include <math.h>

// ---------- sizes ----------
#define NB      512      // graphs
#define NPG     256      // nodes per graph
#define NN      (NB*NPG) // nodes
#define LL      4
#define DD      128
#define PSD     512
#define RD      64

typedef __bf16 bf16x16 __attribute__((ext_vector_type(16)));
typedef __bf16 bf16x8  __attribute__((ext_vector_type(8)));
typedef float  v8f     __attribute__((ext_vector_type(8)));

// ---------------------------------------------------------------------------
// Kernel 1: transpose + convert gate weights to bf16 [128][K] (B^T layout)
//   Bt0: K=576  from A_w  [576,128]
//   Bt1: K=640  from C_w rows {0..511, 1536..1599, 1600..1663}
//   Bt2: K=640  from E_w rows (same mapping)
// ---------------------------------------------------------------------------
__global__ void prep_weights(const float* __restrict__ A_w,
                             const float* __restrict__ C_w,
                             const float* __restrict__ E_w,
                             __bf16* __restrict__ Bt0,
                             __bf16* __restrict__ Bt1,
                             __bf16* __restrict__ Bt2) {
    int idx = blockIdx.x * 256 + threadIdx.x;
    if (idx < 128 * 576) {
        int n = idx / 576, k = idx % 576;
        Bt0[idx] = (__bf16)A_w[k * 128 + n];
    }
    if (idx < 128 * 640) {
        int n = idx / 640, k = idx % 640;
        int kk = (k < 512) ? k : (k < 576 ? (1536 + k - 512) : (1600 + k - 576));
        Bt1[idx] = (__bf16)C_w[kk * 128 + n];
        Bt2[idx] = (__bf16)E_w[kk * 128 + n];
    }
}

// ---------------------------------------------------------------------------
// Kernel 2: per-graph constant vectors (hb/tb parts of gamma/delta pre-acts)
//   cg[g][n] = sum_k hb[k]*C_w[512+k][n] + tb[k]*C_w[1024+k][n]
//   cd[g][n] = sum_k tb[k]*E_w[512+k][n] + hb[k]*E_w[1024+k][n]
// ---------------------------------------------------------------------------
__global__ __launch_bounds__(128)
void prep_consts(const float* __restrict__ flat,
                 const int* __restrict__ head_ids,
                 const int* __restrict__ tail_ids,
                 const float* __restrict__ C_w,
                 const float* __restrict__ E_w,
                 float* __restrict__ cg, float* __restrict__ cd) {
    int g = blockIdx.x, n = threadIdx.x;
    const float* hb = flat + (size_t)head_ids[g] * PSD;
    const float* tb = flat + (size_t)tail_ids[g] * PSD;
    float a = 0.f, b = 0.f;
    for (int k = 0; k < 512; ++k) {
        float h = hb[k], t = tb[k];
        a += h * C_w[(512 + k) * 128 + n] + t * C_w[(1024 + k) * 128 + n];
        b += t * E_w[(512 + k) * 128 + n] + h * E_w[(1024 + k) * 128 + n];
    }
    cg[g * 128 + n] = a;
    cd[g * 128 + n] = b;
}

// ---------------------------------------------------------------------------
// A-fragment loader: p already includes this lane's half*8 offset.
// elems 0..7 <- p[0..7] (K = kbase+0..7), elems 8..15 <- p[16..23] (K += 16)
// (matches the 16-bit A-matrix VGPR layout for 16x16x32 WMMA, wave32)
// ---------------------------------------------------------------------------
__device__ __forceinline__ bf16x16 load_a_frag(const float* __restrict__ p) {
    float4 q0 = ((const float4*)p)[0];
    float4 q1 = ((const float4*)p)[1];
    float4 q2 = ((const float4*)(p + 16))[0];
    float4 q3 = ((const float4*)(p + 16))[1];
    bf16x16 a;
    a[0]  = (__bf16)q0.x; a[1]  = (__bf16)q0.y; a[2]  = (__bf16)q0.z; a[3]  = (__bf16)q0.w;
    a[4]  = (__bf16)q1.x; a[5]  = (__bf16)q1.y; a[6]  = (__bf16)q1.z; a[7]  = (__bf16)q1.w;
    a[8]  = (__bf16)q2.x; a[9]  = (__bf16)q2.y; a[10] = (__bf16)q2.z; a[11] = (__bf16)q2.w;
    a[12] = (__bf16)q3.x; a[13] = (__bf16)q3.y; a[14] = (__bf16)q3.z; a[15] = (__bf16)q3.w;
    return a;
}

// One K-step: 8 WMMAs across the 128 output columns.
// kofs = kcol + half*8 (element offset into the bf16 Bt row of length K).
__device__ __forceinline__ void wmma_step8(bf16x16 a,
                                           const __bf16* __restrict__ Bt,
                                           int K, int kofs, int lr,
                                           v8f* __restrict__ acc) {
#pragma unroll
    for (int nt = 0; nt < 8; ++nt) {
        const __bf16* bp = Bt + (size_t)(nt * 16 + lr) * K + kofs;
        bf16x8 b0 = *(const bf16x8*)bp;
        bf16x8 b1 = *(const bf16x8*)(bp + 16);
        bf16x16 b;
#pragma unroll
        for (int i = 0; i < 8; ++i) { b[i] = b0[i]; b[8 + i] = b1[i]; }
        acc[nt] = __builtin_amdgcn_wmma_f32_16x16x32_bf16(
            false, a, false, b, (short)0, acc[nt], false, false);
    }
}

// ---------------------------------------------------------------------------
// Kernel 3: per-graph fused gate GEMMs + weighted segment means (WMMA bf16)
//   block = 1 graph, 128 threads = 4 waves, each wave owns 4 M-tiles of 16.
// ---------------------------------------------------------------------------
__global__ __launch_bounds__(128)
void node_gate_kernel(const float* __restrict__ flat,
                      const float* __restrict__ head_sister,
                      const float* __restrict__ tail_sister,
                      const int* __restrict__ t_label,
                      const int* __restrict__ hst,
                      const int* __restrict__ tst,
                      const float* __restrict__ rel_emb,
                      const float* __restrict__ she,
                      const float* __restrict__ ste,
                      const __bf16* __restrict__ Bt0,
                      const __bf16* __restrict__ Bt1,
                      const __bf16* __restrict__ Bt2,
                      const float* __restrict__ A_b,
                      const float* __restrict__ C_b,
                      const float* __restrict__ E_b,
                      const float* __restrict__ Bw, const float* __restrict__ Bb,
                      const float* __restrict__ Dw, const float* __restrict__ Db,
                      const float* __restrict__ Gw, const float* __restrict__ Gb,
                      const float* __restrict__ cg, const float* __restrict__ cd,
                      float* __restrict__ g_out,
                      float* __restrict__ ghs,
                      float* __restrict__ gts) {
    __shared__ float h[4][16][128];        // per-wave post-ReLU hidden tile
    __shared__ float gates[4][16][12];     // per-wave: [node][w*4+l]
    __shared__ float num[3][4][128];       // segment-sum numerators
    __shared__ float den[3][4];            // segment-sum denominators

    const int g    = blockIdx.x;
    const int tid  = threadIdx.x;
    const int wave = tid >> 5;
    const int lane = tid & 31;
    const int half = lane >> 4;
    const int lr   = lane & 15;
    const int base = g * NPG;

    for (int i = tid; i < 3 * 4 * 128; i += 128) ((float*)num)[i] = 0.f;
    if (tid < 12) ((float*)den)[tid] = 0.f;
    __syncthreads();

    for (int ti = 0; ti < 4; ++ti) {
        const int tile     = wave + ti * 4;        // 0..15
        const int node_row = base + tile * 16 + lr;
        // loop-invariant per-lane gather bases (hoisted out of all K loops)
        const float* arow = flat + (size_t)node_row * PSD + half * 8;
        const float* seg_rel = rel_emb + t_label[node_row] * RD + half * 8;
        const float* seg_hs  = she + hst[node_row] * RD + half * 8;
        const float* seg_ht  = she + t_label[node_row] * RD + half * 8;
        const float* seg_ts  = ste + tst[node_row] * RD + half * 8;
        const float* seg_tt  = ste + t_label[node_row] * RD + half * 8;

        for (int w = 0; w < 3; ++w) {
            const __bf16* Bt = (w == 0) ? Bt0 : ((w == 1) ? Bt1 : Bt2);
            const int K      = (w == 0) ? 576 : 640;
            const float* seg0 = (w == 0) ? seg_rel : ((w == 1) ? seg_hs : seg_ts);
            const float* seg1 = (w == 1) ? seg_ht : seg_tt;   // unused for w==0
            const int nseg    = (w == 0) ? 1 : 2;

            v8f acc[8] = {};

            // ---- main K = 512: branch-free stream of node features ----
            for (int s = 0; s < 16; ++s) {
                bf16x16 a = load_a_frag(arow + s * 32);
                wmma_step8(a, Bt, K, s * 32 + half * 8, lr, acc);
            }
            // ---- embedding tail: 1 or 2 segments of K=64 (2 steps each) ----
            for (int t = 0; t < nseg; ++t) {
                const float* sp = (t == 0) ? seg0 : seg1;
#pragma unroll
                for (int s2 = 0; s2 < 2; ++s2) {
                    bf16x16 a = load_a_frag(sp + s2 * 32);
                    wmma_step8(a, Bt, K, 512 + t * 64 + s2 * 32 + half * 8, lr, acc);
                }
            }

            // ---- epilogue: bias + per-graph const, ReLU, stash to LDS ----
            const float* bias = (w == 0) ? A_b : ((w == 1) ? C_b : E_b);
#pragma unroll
            for (int nt = 0; nt < 8; ++nt) {
                const int n = nt * 16 + lr;
                float addv = bias[n];
                if (w == 1) addv += cg[g * 128 + n];
                else if (w == 2) addv += cd[g * 128 + n];
#pragma unroll
                for (int j = 0; j < 8; ++j) {
                    float v = acc[nt][j] + addv;
                    h[wave][half * 8 + j][n] = v > 0.f ? v : 0.f;
                }
            }
            __syncthreads();

            // ---- tiny [128]x[128,4] matmul + sigmoid, lanes 0..15 own rows ----
            if (lane < 16) {
                const int r = lane;
                const float* Wsm = (w == 0) ? Bw : ((w == 1) ? Dw : Gw);
                const float* bsm = (w == 0) ? Bb : ((w == 1) ? Db : Gb);
                float pre0 = bsm[0], pre1 = bsm[1], pre2 = bsm[2], pre3 = bsm[3];
                for (int n = 0; n < 128; ++n) {
                    float hv = h[wave][r][n];
                    pre0 += hv * Wsm[n * 4 + 0];
                    pre1 += hv * Wsm[n * 4 + 1];
                    pre2 += hv * Wsm[n * 4 + 2];
                    pre3 += hv * Wsm[n * 4 + 3];
                }
                gates[wave][r][w * 4 + 0] = 1.f / (1.f + __expf(-pre0));
                gates[wave][r][w * 4 + 1] = 1.f / (1.f + __expf(-pre1));
                gates[wave][r][w * 4 + 2] = 1.f / (1.f + __expf(-pre2));
                gates[wave][r][w * 4 + 3] = 1.f / (1.f + __expf(-pre3));
            }
            __syncthreads();
        }

        // ---- weighted segment-mean accumulation for this tile ----
        for (int r = 0; r < 16; ++r) {
            const int node = base + tile * 16 + r;
#pragma unroll
            for (int l = 0; l < 4; ++l) {
                const float wb = gates[wave][r][0 * 4 + l];
                const float wg = gates[wave][r][1 * 4 + l] * head_sister[node * 4 + l];
                const float wd = gates[wave][r][2 * 4 + l] * tail_sister[node * 4 + l];
                if (lane == l) {           // spread den updates over 4 lanes
                    atomicAdd(&den[0][l], wb);
                    atomicAdd(&den[1][l], wg);
                    atomicAdd(&den[2][l], wd);
                }
#pragma unroll
                for (int i = 0; i < 4; ++i) {
                    const int d = lane + 32 * i;
                    const float x = flat[(size_t)node * PSD + l * 128 + d];
                    atomicAdd(&num[0][l][d], wb * x);
                    atomicAdd(&num[1][l][d], wg * x);
                    atomicAdd(&num[2][l][d], wd * x);
                }
            }
        }
    }
    __syncthreads();

    for (int e = tid; e < 512; e += 128) {
        const int l = e >> 7, d = e & 127;
        g_out[g * 512 + e] = num[0][l][d] / den[0][l];
        ghs[g * 512 + e]   = num[1][l][d] / den[1][l];
        gts[g * 512 + e]   = num[2][l][d] / den[2][l];
    }
}

// ---------------------------------------------------------------------------
// Kernel 4: per-graph head: ht MLP (3200->512), fc (2112->16), out (16->1)
// ---------------------------------------------------------------------------
__global__ __launch_bounds__(256)
void head_kernel(const float* __restrict__ flat,
                 const int* __restrict__ head_ids,
                 const int* __restrict__ tail_ids,
                 const int* __restrict__ rel_labels,
                 const float* __restrict__ rel_emb,
                 const float* __restrict__ she,
                 const float* __restrict__ ste,
                 const float* __restrict__ g_out,
                 const float* __restrict__ ghs,
                 const float* __restrict__ gts,
                 const float* __restrict__ ht_w, const float* __restrict__ ht_b,
                 const float* __restrict__ fc_w, const float* __restrict__ fc_b,
                 const float* __restrict__ out_w, const float* __restrict__ out_b,
                 float* __restrict__ out) {
    __shared__ float ght[3200];
    __shared__ float grep[2112];
    __shared__ float fcv[16];

    const int g = blockIdx.x, tid = threadIdx.x;
    const float* hN = flat + (size_t)head_ids[g] * PSD;
    const float* tN = flat + (size_t)tail_ids[g] * PSD;
    const float* GS = ghs + g * PSD;
    const float* GT = gts + g * PSD;
    const float* GO = g_out + g * PSD;
    const int rl = rel_labels[g];

    for (int i = tid; i < 512; i += 256) {
        float hh = hN[i], tt = tN[i], gs = GS[i], gt = GT[i];
        ght[i]        = gs * hh;
        ght[512 + i]  = gt * tt;
        ght[1024 + i] = gs;
        ght[1536 + i] = gt;
        ght[2048 + i] = hh;
        ght[2560 + i] = tt;
        grep[i]        = GO[i];
        grep[512 + i]  = hh;
        grep[1024 + i] = tt;
    }
    if (tid < 64) {
        ght[3072 + tid] = she[rl * RD + tid];
        ght[3136 + tid] = ste[rl * RD + tid];
        grep[2048 + tid] = rel_emb[rl * RD + tid];
    }
    __syncthreads();

    for (int j = tid; j < 512; j += 256) {
        float acc = ht_b[j];
        for (int k = 0; k < 3200; ++k) acc += ght[k] * ht_w[(size_t)k * 512 + j];
        grep[1536 + j] = acc > 0.f ? acc : 0.f;
    }
    __syncthreads();

    if (tid < 16) {
        float acc = fc_b[tid];
        for (int k = 0; k < 2112; ++k) acc += grep[k] * fc_w[k * 16 + tid];
        fcv[tid] = acc > 0.f ? acc : 0.f;
    }
    __syncthreads();

    if (tid == 0) {
        float acc = out_b[0];
        for (int j = 0; j < 16; ++j) acc += fcv[j] * out_w[j];
        out[g] = acc;
    }
}

// ---------------------------------------------------------------------------
extern "C" void kernel_launch(void* const* d_in, const int* in_sizes, int n_in,
                              void* d_out, int out_size, void* d_ws, size_t ws_size,
                              hipStream_t stream) {
    const float* node_repr   = (const float*)d_in[0];
    const float* head_sister = (const float*)d_in[1];
    const float* tail_sister = (const float*)d_in[2];
    const int*   t_label     = (const int*)d_in[3];
    const int*   hst         = (const int*)d_in[4];
    const int*   tst         = (const int*)d_in[5];
    /* d_in[6] graph_ids: implicit (node/NPG) */
    const int*   head_ids    = (const int*)d_in[7];
    const int*   tail_ids    = (const int*)d_in[8];
    const int*   rel_labels  = (const int*)d_in[9];
    const float* rel_emb     = (const float*)d_in[10];
    const float* she         = (const float*)d_in[11];
    const float* ste         = (const float*)d_in[12];
    const float* A_w  = (const float*)d_in[13];
    const float* A_b  = (const float*)d_in[14];
    const float* B_w  = (const float*)d_in[15];
    const float* B_b  = (const float*)d_in[16];
    const float* C_w  = (const float*)d_in[17];
    const float* C_b  = (const float*)d_in[18];
    const float* Dm_w = (const float*)d_in[19];
    const float* Dm_b = (const float*)d_in[20];
    const float* E_w  = (const float*)d_in[21];
    const float* E_b  = (const float*)d_in[22];
    const float* G_w  = (const float*)d_in[23];
    const float* G_b  = (const float*)d_in[24];
    const float* ht_w = (const float*)d_in[25];
    const float* ht_b = (const float*)d_in[26];
    const float* fc_w = (const float*)d_in[27];
    const float* fc_b = (const float*)d_in[28];
    const float* out_w = (const float*)d_in[29];
    const float* out_b = (const float*)d_in[30];

    char* ws = (char*)d_ws;
    size_t off = 0;
    __bf16* Bt0 = (__bf16*)(ws + off); off += (size_t)128 * 576 * 2;  // 147456
    __bf16* Bt1 = (__bf16*)(ws + off); off += (size_t)128 * 640 * 2;  // 163840
    __bf16* Bt2 = (__bf16*)(ws + off); off += (size_t)128 * 640 * 2;  // 163840
    float* cg    = (float*)(ws + off); off += (size_t)NB * 128 * 4;
    float* cd    = (float*)(ws + off); off += (size_t)NB * 128 * 4;
    float* g_out = (float*)(ws + off); off += (size_t)NB * PSD * 4;
    float* w_ghs = (float*)(ws + off); off += (size_t)NB * PSD * 4;
    float* w_gts = (float*)(ws + off); off += (size_t)NB * PSD * 4;

    prep_weights<<<(128 * 640 + 255) / 256, 256, 0, stream>>>(A_w, C_w, E_w, Bt0, Bt1, Bt2);
    prep_consts<<<NB, 128, 0, stream>>>(node_repr, head_ids, tail_ids, C_w, E_w, cg, cd);
    node_gate_kernel<<<NB, 128, 0, stream>>>(
        node_repr, head_sister, tail_sister, t_label, hst, tst,
        rel_emb, she, ste, Bt0, Bt1, Bt2,
        A_b, C_b, E_b, B_w, B_b, Dm_w, Dm_b, G_w, G_b,
        cg, cd, g_out, w_ghs, w_gts);
    head_kernel<<<NB, 256, 0, stream>>>(
        node_repr, head_ids, tail_ids, rel_labels, rel_emb, she, ste,
        g_out, w_ghs, w_gts, ht_w, ht_b, fc_w, fc_b, out_w, out_b,
        (float*)d_out);
}
